// BASE_6442450944602
// MI455X (gfx1250) — matrix operand order repaired and checked by reading, not compiled, not run on hardware
//
#include <hip/hip_runtime.h>

// ---------------------------------------------------------------------------
// MI455X (gfx1250, wave32) implementation.
// Stages:
//   1. se_reduce   : s[b,c] = mean_{yx} x          (wave-per-row, shfl_xor reduce)
//   2. se_mlp      : y[b,c] = sigmoid(relu(s W1^T+b1) W2^T + b2)
//   3. f2bf        : gus, down_w -> bf16
//   4. scale_tr    : Xb[b][s][c] = bf16(x[b,c,s] * y[b,c])   (LDS 32x32 transpose)
//   5. gemm (WMMA) : cat[b][0:512][t]  <- gus[1024x1024] @ Xb[b] (bf16 out)
//   6. csa         : cat[b][512:1024][t] <- patch softmax attention (bf16 out)
//   7. gemm (WMMA) : z[b] = down_w[512x1024] @ cat[b][1024x1024] (f32 out)
//   8. instnorm    : per (b,o) mean/var over 1024 + LeakyReLU(0.2) -> d_out
// The reshape/concat trickery of the reference collapses: gus_out flat [p][c]
// followed by g2 flat [s][c] IS the row-major [1024 ch][1024 sp] B matrix of
// the down GEMM.
// GEMM: 128x128 block tile, double-buffered LDS (one barrier / K-step),
// all fragment gathers hoisted ahead of the 8 back-to-back WMMAs so no
// WAR hazards (v_nop) land between v_wmma instructions.
// ---------------------------------------------------------------------------

#define DEVINL __device__ __forceinline__

typedef __attribute__((ext_vector_type(16))) __bf16 v16bf;
typedef __attribute__((ext_vector_type(8)))  float  v8f;

static constexpr int BATCH = 16;
static constexpr int C     = 512;
static constexpr int S     = 1024;   // H*W = 32*32
static constexpr int CR    = 32;     // C / RED

DEVINL unsigned short f2bf(float f) {
  unsigned int u = __float_as_uint(f);
  u += 0x7FFFu + ((u >> 16) & 1u);          // round to nearest even
  return (unsigned short)(u >> 16);
}
DEVINL float bf2f(unsigned short h) {
  return __uint_as_float((unsigned int)h << 16);
}
DEVINL float wave_sum(float v) {
  #pragma unroll
  for (int off = 16; off; off >>= 1) v += __shfl_xor(v, off, 32);
  return v;
}
DEVINL float sigmoidf_(float x) { return 1.0f / (1.0f + __expf(-x)); }

// ---------------------------------------------------------------- stage 1 ---
__global__ __launch_bounds__(256) void se_reduce_kernel(
    const float* __restrict__ x, float* __restrict__ smean) {
  int row  = blockIdx.x * 8 + (threadIdx.x >> 5);   // (b*C + c), 8192 rows
  int lane = threadIdx.x & 31;
  const float* p = x + (size_t)row * S;
  float acc = 0.f;
  #pragma unroll
  for (int i = 0; i < S / 32; ++i) acc += p[lane + 32 * i];
  acc = wave_sum(acc);
  if (lane == 0) smean[row] = acc * (1.0f / (float)S);
}

// ---------------------------------------------------------------- stage 2 ---
__global__ __launch_bounds__(256) void se_mlp_kernel(
    const float* __restrict__ smean,
    const float* __restrict__ w1, const float* __restrict__ b1,
    const float* __restrict__ w2, const float* __restrict__ b2,
    float* __restrict__ yscale) {
  __shared__ float sv[C];
  __shared__ float hv[CR];
  int b = blockIdx.x, t = threadIdx.x;
  sv[t]       = smean[b * C + t];
  sv[t + 256] = smean[b * C + t + 256];
  __syncthreads();
  if (t < CR) {
    float a = b1[t];
    const float* wr = w1 + t * C;
    for (int c = 0; c < C; ++c) a += sv[c] * wr[c];
    hv[t] = fmaxf(a, 0.f);
  }
  __syncthreads();
  #pragma unroll
  for (int r = 0; r < 2; ++r) {
    int c = t + r * 256;
    float a = b2[c];
    const float* wr = w2 + c * CR;
    #pragma unroll
    for (int j = 0; j < CR; ++j) a += hv[j] * wr[j];
    yscale[b * C + c] = sigmoidf_(a);
  }
}

// ---------------------------------------------------------------- stage 3 ---
__global__ __launch_bounds__(256) void f2bf_kernel(
    const float* __restrict__ in, unsigned short* __restrict__ out, int n) {
  int i = blockIdx.x * 256 + threadIdx.x;
  if (i < n) out[i] = f2bf(in[i]);
}

// ---------------------------------------------------------------- stage 4 ---
// Xb[b][s][c] = bf16( x[b][c][s] * y[b][c] )   via 32x32 LDS transpose tiles
__global__ __launch_bounds__(256) void scale_transpose_kernel(
    const float* __restrict__ x, const float* __restrict__ yscale,
    unsigned short* __restrict__ Xb) {
  __shared__ unsigned short tile[32][33];
  int b = blockIdx.z, c0 = blockIdx.y * 32, s0 = blockIdx.x * 32;
  int tx = threadIdx.x & 31, ty = threadIdx.x >> 5;
  #pragma unroll
  for (int r = 0; r < 4; ++r) {
    int cc = c0 + ty + 8 * r;
    float v = x[((size_t)(b * C + cc)) * S + s0 + tx] * yscale[b * C + cc];
    tile[ty + 8 * r][tx] = f2bf(v);
  }
  __syncthreads();
  #pragma unroll
  for (int r = 0; r < 4; ++r) {
    int ss = s0 + ty + 8 * r;
    Xb[((size_t)(b * S + ss)) * C + c0 + tx] = tile[tx][ty + 8 * r];
  }
}

// ------------------------------------------------------------ WMMA GEMM ----
// C[b] = A[MxK] (shared) @ B[b][KxN], all bf16 row-major, f32 accumulate.
// Block tile 128x128, 8 waves (2x4), each wave 64x32 = 4x2 WMMA tiles.
// Double-buffered LDS: one barrier per K-step; next tile's global loads
// overlap the current tile's WMMAs. All fragments are gathered into
// registers before the WMMA burst.
template<int OUT_BF16>
__global__ __launch_bounds__(256) void gemm_bf16_128x128(
    const unsigned short* __restrict__ A,
    const unsigned short* __restrict__ Bmat, long strideB,
    void* __restrict__ Cout, long strideC,
    int M, int N, int K) {
  constexpr int AS_STR   = 34;               // pad: conflict-free gathers
  constexpr int TILE_U16 = 128 * AS_STR;
  __shared__ unsigned short As[2 * TILE_U16];   // [buf][m][k]
  __shared__ unsigned short Bs[2 * TILE_U16];   // [buf][n][k] (transposed tile)

  const int bId = blockIdx.z;
  const unsigned short* Bb = Bmat + (size_t)bId * strideB;
  const int bm = blockIdx.y * 128;
  const int bn = blockIdx.x * 128;
  const int tid  = threadIdx.x;
  const int wv   = tid >> 5, lane = tid & 31;
  const int wm   = wv >> 2,  wn   = wv & 3;
  const int lr   = lane & 15;
  const int lhalf = lane >> 4;                 // 0 or 1

  const v8f vzero = {0.f, 0.f, 0.f, 0.f, 0.f, 0.f, 0.f, 0.f};
  v8f acc[4][2];
  #pragma unroll
  for (int mt = 0; mt < 4; ++mt)
    #pragma unroll
    for (int nt = 0; nt < 2; ++nt) acc[mt][nt] = vzero;

  const int arow = tid >> 1, ahalf = tid & 1;  // A: 128 rows x 2 halves
  const int bkp  = tid >> 4;                   // B: k-pair 0..15 -> k = 2*bkp
  const int bnc  = (tid & 15) * 8;             // B: chunk of 8 columns

  // Stage one 128x32 A tile and 32x128 B tile (transposed) into LDS buffer.
  auto loadTiles = [&](int kk, int buf) {
    unsigned short* Asb = As + buf * TILE_U16;
    unsigned short* Bsb = Bs + buf * TILE_U16;
    {   // As[m][k] = A[bm+m][kk+k] : 2 x 16B per thread, dword LDS stores
      const uint4* g = reinterpret_cast<const uint4*>(
          A + (size_t)(bm + arow) * K + kk + ahalf * 16);
      uint4 p = g[0], q = g[1];
      unsigned int* d =
          reinterpret_cast<unsigned int*>(&Asb[arow * AS_STR + ahalf * 16]);
      d[0] = p.x; d[1] = p.y; d[2] = p.z; d[3] = p.w;
      d[4] = q.x; d[5] = q.y; d[6] = q.z; d[7] = q.w;
    }
    {   // Bs[n][k] = B[kk+k][bn+n]; pack k-pair into one dword store
      const uint4* g0 = reinterpret_cast<const uint4*>(
          Bb + (size_t)(kk + 2 * bkp) * N + bn + bnc);
      const uint4* g1 = reinterpret_cast<const uint4*>(
          Bb + (size_t)(kk + 2 * bkp + 1) * N + bn + bnc);
      union { uint4 q; unsigned short h[8]; } r0, r1;
      r0.q = g0[0]; r1.q = g1[0];
      unsigned int* Bw = reinterpret_cast<unsigned int*>(Bsb);
      #pragma unroll
      for (int j = 0; j < 8; ++j) {
        unsigned int v = (unsigned int)r0.h[j] | ((unsigned int)r1.h[j] << 16);
        Bw[(bnc + j) * (AS_STR / 2) + bkp] = v;
      }
    }
  };

  loadTiles(0, 0);
  __syncthreads();

  union Frag { v16bf v; unsigned int u[8]; };

  for (int kk = 0; kk < K; kk += 32) {
    const int cur = (kk >> 5) & 1;
    if (kk + 32 < K) loadTiles(kk + 32, cur ^ 1);
    if (kk + 64 < K) {   // gfx1250 global_prefetch_b8, two tiles ahead
      __builtin_prefetch(A + (size_t)(bm + arow) * K + kk + 64, 0, 3);
      __builtin_prefetch(Bb + (size_t)(kk + 64 + 2 * bkp) * N + bn + bnc, 0, 3);
    }
    const unsigned short* Asb = As + cur * TILE_U16;
    const unsigned short* Bsb = Bs + cur * TILE_U16;

    // ---- gather ALL fragments first (distinct regs, no WAR vs WMMA) ----
    // B frags: lane<16 -> col=lr, K=0..15; lane>=16 -> col=lr, K=16..31
    Frag bfr[2];
    #pragma unroll
    for (int nt = 0; nt < 2; ++nt) {
      int col = wn * 32 + nt * 16 + lr;
      #pragma unroll
      for (int v = 0; v < 8; ++v) {
        int k0 = 2 * v + lhalf * 16;
        bfr[nt].u[v] =
            *reinterpret_cast<const unsigned int*>(&Bsb[col * AS_STR + k0]);
      }
    }
    // A frags: row=lr; K=0..15 (lanes 0-15) vs K=16..31 (lanes 16-31)
    Frag af[4];
    #pragma unroll
    for (int mt = 0; mt < 4; ++mt) {
      int row = wm * 64 + mt * 16 + lr;
      #pragma unroll
      for (int v = 0; v < 8; ++v) {
        int k0 = 2 * v + lhalf * 16;
        af[mt].u[v] =
            *reinterpret_cast<const unsigned int*>(&Asb[row * AS_STR + k0]);
      }
    }
    // ---- 8 back-to-back WMMAs ----
    #pragma unroll
    for (int mt = 0; mt < 4; ++mt) {
      #pragma unroll
      for (int nt = 0; nt < 2; ++nt) {
        acc[mt][nt] = __builtin_amdgcn_wmma_f32_16x16x32_bf16(
            false, af[mt].v, false, bfr[nt].v, (short)0, acc[mt][nt],
            false, false);
      }
    }
    __syncthreads();
  }

  // epilogue: D layout — lanes 0-15: M=r, N=lane; lanes 16-31: M=8+r, N=lane-16
  #pragma unroll
  for (int mt = 0; mt < 4; ++mt) {
    #pragma unroll
    for (int nt = 0; nt < 2; ++nt) {
      #pragma unroll
      for (int r = 0; r < 8; ++r) {
        int gm = bm + wm * 64 + mt * 16 + lhalf * 8 + r;
        int gn = bn + wn * 32 + nt * 16 + lr;
        float val = acc[mt][nt][r];
        if (OUT_BF16) {
          ((unsigned short*)Cout + (size_t)bId * strideC)[(size_t)gm * N + gn] =
              f2bf(val);
        } else {
          ((float*)Cout + (size_t)bId * strideC)[(size_t)gm * N + gn] = val;
        }
      }
    }
  }
}

// ---------------------------------------------------------------- stage 6 ---
// One wave per spatial position: 9 channel-dots (len 512), softmax over 9,
// weighted sum of neighbor channel vectors. Zero-padded borders contribute 0
// (pad is applied AFTER sigmoid in the reference, so padded sig == 0).
__global__ __launch_bounds__(256) void csa_kernel(
    const unsigned short* __restrict__ Xb, unsigned short* __restrict__ cat) {
  int wv = threadIdx.x >> 5, lane = threadIdx.x & 31;
  int pos = blockIdx.x * 8 + wv;                 // 0 .. BATCH*S-1
  int b = pos >> 10, s = pos & 1023;
  int y = s >> 5, x = s & 31;
  const unsigned short* base = Xb + ((size_t)b * S) * C;

  float sigc[16];
  {
    const unsigned short* p = base + (size_t)s * C + lane * 16;
    #pragma unroll
    for (int i = 0; i < 16; ++i) sigc[i] = sigmoidf_(bf2f(p[i]));
  }
  float lg[9];
  #pragma unroll
  for (int k = 0; k < 9; ++k) {
    int yy = y + k / 3 - 1, xx = x + k % 3 - 1;
    float part = 0.f;
    if ((unsigned)yy < 32u && (unsigned)xx < 32u) {
      const unsigned short* p = base + (size_t)(yy * 32 + xx) * C + lane * 16;
      #pragma unroll
      for (int i = 0; i < 16; ++i) part += sigc[i] * sigmoidf_(bf2f(p[i]));
    }
    lg[k] = wave_sum(part) * (1.0f / (float)C);
  }
  float m = lg[0];
  #pragma unroll
  for (int k = 1; k < 9; ++k) m = fmaxf(m, lg[k]);
  float e[9], sum = 0.f;
  #pragma unroll
  for (int k = 0; k < 9; ++k) { e[k] = __expf(lg[k] - m); sum += e[k]; }
  float inv = 1.0f / sum;

  float out[16];
  #pragma unroll
  for (int i = 0; i < 16; ++i) out[i] = 0.f;
  #pragma unroll
  for (int k = 0; k < 9; ++k) {
    int yy = y + k / 3 - 1, xx = x + k % 3 - 1;
    if ((unsigned)yy < 32u && (unsigned)xx < 32u) {
      float w = e[k] * inv;
      const unsigned short* p = base + (size_t)(yy * 32 + xx) * C + lane * 16;
      #pragma unroll
      for (int i = 0; i < 16; ++i) out[i] += w * bf2f(p[i]);
    }
  }
  unsigned short* dst =
      cat + (size_t)b * (2 * C * S) + (size_t)C * S + (size_t)s * C + lane * 16;
  #pragma unroll
  for (int i = 0; i < 16; ++i) dst[i] = f2bf(out[i]);
}

// ---------------------------------------------------------------- stage 8 ---
__global__ __launch_bounds__(256) void instnorm_kernel(
    const float* __restrict__ z, float* __restrict__ out) {
  int row  = blockIdx.x * 8 + (threadIdx.x >> 5);   // (b*C + o)
  int lane = threadIdx.x & 31;
  const float* p = z + (size_t)row * S;
  float s1 = 0.f, s2 = 0.f;
  #pragma unroll
  for (int i = 0; i < S / 32; ++i) {
    float v = p[lane + 32 * i];
    s1 += v; s2 += v * v;
  }
  s1 = wave_sum(s1); s2 = wave_sum(s2);
  float mean = s1 * (1.0f / (float)S);
  float var  = s2 * (1.0f / (float)S) - mean * mean;
  float rs   = rsqrtf(var + 1e-5f);
  float* o = out + (size_t)row * S;
  #pragma unroll
  for (int i = 0; i < S / 32; ++i) {
    float v = (p[lane + 32 * i] - mean) * rs;
    o[lane + 32 * i] = (v >= 0.f) ? v : 0.2f * v;
  }
}

// ---------------------------------------------------------------------------
extern "C" void kernel_launch(void* const* d_in, const int* in_sizes, int n_in,
                              void* d_out, int out_size, void* d_ws,
                              size_t ws_size, hipStream_t stream) {
  const float* x   = (const float*)d_in[0];
  const float* w1  = (const float*)d_in[1];
  const float* b1  = (const float*)d_in[2];
  const float* w2  = (const float*)d_in[3];
  const float* b2  = (const float*)d_in[4];
  const float* dw  = (const float*)d_in[5];
  const float* gus = (const float*)d_in[6];

  char* ws = (char*)d_ws;
  size_t off = 0;
  auto alloc = [&](size_t bytes) -> void* {
    void* p = ws + off;
    off = (off + bytes + 255) & ~(size_t)255;
    return p;
  };
  float*          smean  = (float*)alloc((size_t)BATCH * C * 4);
  float*          yscale = (float*)alloc((size_t)BATCH * C * 4);
  unsigned short* gus_bf = (unsigned short*)alloc((size_t)S * S * 2);
  unsigned short* dw_bf  = (unsigned short*)alloc((size_t)C * 2 * C * 2);
  unsigned short* Xb     = (unsigned short*)alloc((size_t)BATCH * S * C * 2);
  unsigned short* cat    = (unsigned short*)alloc((size_t)BATCH * 2 * C * S * 2);
  float*          z      = (float*)alloc((size_t)BATCH * C * S * 4);

  se_reduce_kernel<<<BATCH * C / 8, 256, 0, stream>>>(x, smean);
  se_mlp_kernel<<<BATCH, 256, 0, stream>>>(smean, w1, b1, w2, b2, yscale);
  f2bf_kernel<<<(S * S) / 256, 256, 0, stream>>>(gus, gus_bf, S * S);
  f2bf_kernel<<<(C * 2 * C) / 256, 256, 0, stream>>>(dw, dw_bf, C * 2 * C);
  scale_transpose_kernel<<<dim3(S / 32, C / 32, BATCH), 256, 0, stream>>>(
      x, yscale, Xb);
  // gus_out[b][p][c] = gus[1024x1024] @ Xb[b][1024x512]  -> cat first half
  gemm_bf16_128x128<1><<<dim3(C / 128, S / 128, BATCH), 256, 0, stream>>>(
      gus_bf, Xb, (long)S * C, cat, (long)2 * C * S, S, C, S);
  // g2[b][s][c] -> cat second half
  csa_kernel<<<BATCH * S / 8, 256, 0, stream>>>(Xb, cat);
  // z[b][o][t] = down_w[512x1024] @ cat[b][1024x1024]
  gemm_bf16_128x128<0><<<dim3(S / 128, C / 128, BATCH), 256, 0, stream>>>(
      dw_bf, cat, (long)2 * C * S, z, (long)C * S, C, S, 2 * C);
  instnorm_kernel<<<BATCH * C / 8, 256, 0, stream>>>(z, (float*)d_out);
}
